// Transformer_38646115729644
// MI455X (gfx1250) — matrix operand order repaired and checked by reading, not compiled, not run on hardware
//
#include <hip/hip_runtime.h>
#include <math.h>

#define LAYERS 6
#define BATCH  2
#define SEQ    1024
#define DIM    1024
#define HEADS  16
#define DH     64
#define ROWS   (BATCH * SEQ)   // 2048

typedef _Float16 f16;
typedef _Float16 v16h __attribute__((ext_vector_type(16)));
typedef _Float16 v8h  __attribute__((ext_vector_type(8)));
typedef float    v8f  __attribute__((ext_vector_type(8)));
typedef unsigned int u32x4 __attribute__((ext_vector_type(4)));
typedef int          i32x4 __attribute__((ext_vector_type(4)));
typedef int          i32x8 __attribute__((ext_vector_type(8)));

#if defined(__has_builtin)
#  if __has_builtin(__builtin_amdgcn_tensor_load_to_lds) && \
      __has_builtin(__builtin_amdgcn_s_wait_tensorcnt)
#    define USE_TDM 1
#  endif
#endif
#ifndef USE_TDM
#  define USE_TDM 0
#endif

static __device__ __forceinline__ v16h mk16(v8h lo, v8h hi) {
  return __builtin_shufflevector(lo, hi, 0,1,2,3,4,5,6,7,8,9,10,11,12,13,14,15);
}
static __device__ __forceinline__ v8f wmma_f16(v16h a, v16h b, v8f c) {
  return __builtin_amdgcn_wmma_f32_16x16x32_f16(false, a, false, b, (short)0, c, false, false);
}

#if USE_TDM
static __device__ __forceinline__ unsigned lds_addr_of(const void* p) {
  return (unsigned)(unsigned long long)
         (__attribute__((address_space(3))) const char*)p;
}
// 2-D TDM: copy `rows` x 64 halves from global (row stride = row_stride_elems
// halves) into LDS, DMA-padding 16B after each 128B row -> LDS stride 72 halves.
static __device__ __forceinline__ void tdm_load_2d(const f16* gptr, const void* lds,
                                                   int rows, int row_stride_elems) {
  const unsigned long long ga = (unsigned long long)gptr;
  u32x4 g0;
  g0[0] = 1u;                                                  // count=1 (valid)
  g0[1] = lds_addr_of(lds);                                    // lds_addr
  g0[2] = (unsigned)ga;                                        // global_addr lo
  g0[3] = ((unsigned)(ga >> 32) & 0x01FFFFFFu) | (2u << 30);   // addr hi | type=2
  i32x8 g1;
  g1[0] = (1 << 16)     // data_size = 2 bytes
        | (1 << 20)     // pad_enable
        | (4 << 22)     // pad_interval: 32 DWORDs (128B)
        | (3 << 25);    // pad_amount: 4 DWORDs (16B)
  g1[1] = (64 & 0xFFFF) << 16;                  // tensor_dim0 lo (=64)
  g1[2] = 0 | ((rows & 0xFFFF) << 16);          // tensor_dim0 hi | tensor_dim1 lo
  g1[3] = ((rows >> 16) & 0xFFFF) | (64 << 16); // tensor_dim1 hi | tile_dim0=64
  g1[4] = rows & 0xFFFF;                        // tile_dim1 = rows, tile_dim2=0
  g1[5] = row_stride_elems;                     // tensor_dim0_stride lo
  g1[6] = 0;                                    // stride hi | dim1_stride lo
  g1[7] = 0;
  const i32x4 z4 = {0, 0, 0, 0};
  const i32x8 z8 = {0, 0, 0, 0, 0, 0, 0, 0};
  __builtin_amdgcn_tensor_load_to_lds(g0, g1, z4, z4, z8, 0);
}
#endif

// ---------------------------------------------------------------------------
// Weight prep: fp32 [K][N] -> f16 transposed [N][K], batched over layers (z)
// ---------------------------------------------------------------------------
__global__ __launch_bounds__(256)
void wprep_kernel(const float* __restrict__ in, f16* __restrict__ out, int K, int N) {
  __shared__ f16 tile[32][33];
  const size_t ls = (size_t)K * N;
  const float* ip = in + (size_t)blockIdx.z * ls;
  f16* op = out + (size_t)blockIdx.z * ls;
  const int n0 = blockIdx.x * 32, k0 = blockIdx.y * 32;
  const int tx = threadIdx.x, ty = threadIdx.y;   // 32 x 8
#pragma unroll
  for (int r = 0; r < 4; ++r)
    tile[ty + r * 8][tx] = (f16)ip[(size_t)(k0 + ty + r * 8) * N + n0 + tx];
  __syncthreads();
#pragma unroll
  for (int r = 0; r < 4; ++r)
    op[(size_t)(n0 + ty + r * 8) * K + k0 + tx] = tile[tx][ty + r * 8];
}

// ---------------------------------------------------------------------------
// LayerNorm over D=1024, one row per 256-thread block; OutT = f16 or float
// ---------------------------------------------------------------------------
template <typename OutT>
__global__ __launch_bounds__(256)
void ln_kernel(const float* __restrict__ x, const float* __restrict__ w,
               const float* __restrict__ b, OutT* __restrict__ out) {
  __shared__ float rs[256], rq[256];
  const int row = blockIdx.x, tid = threadIdx.x;
  const float4 xv = ((const float4*)(x + (size_t)row * DIM))[tid];
  rs[tid] = xv.x + xv.y + xv.z + xv.w;
  rq[tid] = xv.x * xv.x + xv.y * xv.y + xv.z * xv.z + xv.w * xv.w;
  __syncthreads();
  for (int off = 128; off > 0; off >>= 1) {
    if (tid < off) { rs[tid] += rs[tid + off]; rq[tid] += rq[tid + off]; }
    __syncthreads();
  }
  const float mean = rs[0] * (1.0f / DIM);
  const float var  = rq[0] * (1.0f / DIM) - mean * mean;
  const float rinv = rsqrtf(var + 1e-5f);
  const float4 wv = ((const float4*)w)[tid];
  const float4 bv = ((const float4*)b)[tid];
  OutT* orow = out + (size_t)row * DIM + tid * 4;
  orow[0] = (OutT)((xv.x - mean) * rinv * wv.x + bv.x);
  orow[1] = (OutT)((xv.y - mean) * rinv * wv.y + bv.y);
  orow[2] = (OutT)((xv.z - mean) * rinv * wv.z + bv.z);
  orow[3] = (OutT)((xv.w - mean) * rinv * wv.w + bv.w);
}

// ---------------------------------------------------------------------------
// GEMM: out[M,N] = epi(A[M,K](f16) @ Wt[N,K]^T(f16) + bias) (+res)
// 128x64 block tile, BK=64, 256 threads = 8 waves (4x2), 32x32 per wave.
// Tile staging via TDM (double-buffered) when available.
// ---------------------------------------------------------------------------
#define EPI_BIAS 0
#define EPI_GELU 1
#define EPI_RES  2

template <int EPI, typename OutT>
__global__ __launch_bounds__(256)
void gemm_kernel(const f16* __restrict__ A, const f16* __restrict__ Wt,
                 const float* __restrict__ bias, const float* __restrict__ res,
                 OutT* __restrict__ out, int M, int N, int K) {
  (void)M;
  const int tid  = threadIdx.x;
  const int wave = tid >> 5, lane = tid & 31;
  const int hi = lane >> 4, lm = lane & 15;
  const int wm = (wave >> 1) * 32;          // 0,32,64,96
  const int wn = (wave & 1) * 32;           // 0,32
  const int bm = blockIdx.x * 128;
  const int bn = blockIdx.y * 64;

  v8f acc[2][2] = {};
  auto compute_stage = [&](const f16 (*Asr)[72], const f16 (*Bsr)[72]) {
#pragma unroll
    for (int ks = 0; ks < 2; ++ks) {
      const int ka = 8 * hi + 32 * ks;      // A-frag K base (ISA A layout)
      v16h a0 = mk16(*(const v8h*)&Asr[wm + lm][ka],      *(const v8h*)&Asr[wm + lm][ka + 16]);
      v16h a1 = mk16(*(const v8h*)&Asr[wm + 16 + lm][ka], *(const v8h*)&Asr[wm + 16 + lm][ka + 16]);
      const int kbb = 8 * hi + 32 * ks;     // B-frag K base (ISA B layout)
      v16h b0 = mk16(*(const v8h*)&Bsr[wn + lm][kbb],      *(const v8h*)&Bsr[wn + lm][kbb + 16]);
      v16h b1 = mk16(*(const v8h*)&Bsr[wn + 16 + lm][kbb], *(const v8h*)&Bsr[wn + 16 + lm][kbb + 16]);
      acc[0][0] = wmma_f16(a0, b0, acc[0][0]);
      acc[0][1] = wmma_f16(a0, b1, acc[0][1]);
      acc[1][0] = wmma_f16(a1, b0, acc[1][0]);
      acc[1][1] = wmma_f16(a1, b1, acc[1][1]);
    }
  };

#if USE_TDM
  __shared__ __align__(16) f16 As[2][128][72];
  __shared__ __align__(16) f16 Bs[2][64][72];
  const int nk = K >> 6;
  if (wave == 0) {                          // prologue: stage 0 into buffer 0
    tdm_load_2d(&A[(size_t)bm * K], &As[0][0][0], 128, K);
    tdm_load_2d(&Wt[(size_t)bn * K], &Bs[0][0][0], 64, K);
  }
  for (int kb = 0; kb < nk; ++kb) {
    const int p = kb & 1;
    if (wave == 0) {
      if (kb + 1 < nk) {                    // prefetch next stage into other buf
        tdm_load_2d(&A[(size_t)bm * K + (kb + 1) * 64], &As[1 - p][0][0], 128, K);
        tdm_load_2d(&Wt[(size_t)bn * K + (kb + 1) * 64], &Bs[1 - p][0][0], 64, K);
        __builtin_amdgcn_s_wait_tensorcnt(2);  // stage kb done (in-order)
      } else {
        __builtin_amdgcn_s_wait_tensorcnt(0);
      }
    }
    __syncthreads();                        // DMA'd tiles visible to all waves
    compute_stage(As[p], Bs[p]);
    __syncthreads();                        // done reading before buf reuse
  }
#else
  __shared__ __align__(16) f16 As[128][72];
  __shared__ __align__(16) f16 Bs[64][72];
  for (int kb = 0; kb < K; kb += 64) {
#pragma unroll
    for (int t = 0; t < 4; ++t) {           // A tile: 128x64 halves
      const int idx = tid + t * 256;
      const int r = idx >> 3, c8 = (idx & 7) * 8;
      *(v8h*)&As[r][c8] = *(const v8h*)&A[(size_t)(bm + r) * K + kb + c8];
    }
#pragma unroll
    for (int t = 0; t < 2; ++t) {           // B tile: 64x64 halves
      const int idx = tid + t * 256;
      const int r = idx >> 3, c8 = (idx & 7) * 8;
      *(v8h*)&Bs[r][c8] = *(const v8h*)&Wt[(size_t)(bn + r) * K + kb + c8];
    }
    __syncthreads();
    compute_stage(As, Bs);
    __syncthreads();
  }
#endif

#pragma unroll
  for (int i = 0; i < 2; ++i) {
#pragma unroll
    for (int j = 0; j < 2; ++j) {
      const int col = bn + wn + j * 16 + lm;
      const float bj = bias[col];
#pragma unroll
      for (int v = 0; v < 8; ++v) {
        const int row = bm + wm + i * 16 + v + 8 * hi;  // ISA C layout
        float val = acc[i][j][v] + bj;
        if constexpr (EPI == EPI_GELU)
          val = 0.5f * val * (1.0f + erff(val * 0.70710678118654752f));
        if constexpr (EPI == EPI_RES)
          val += res[(size_t)row * N + col];
        out[(size_t)row * N + col] = (OutT)val;
      }
    }
  }
}

// ---------------------------------------------------------------------------
// Flash attention, block-causal (TPB=64): query block qb sees keys [0,(qb+1)*64)
// 128 threads = 4 waves; each wave owns 16 query rows; all matmuls via WMMA.
// ---------------------------------------------------------------------------
__global__ __launch_bounds__(128)
void attn_kernel(const f16* __restrict__ qkv, f16* __restrict__ y) {
  __shared__ __align__(16) f16 Qs[64][72];
  __shared__ __align__(16) f16 Ks[64][72];
  __shared__ __align__(16) f16 Vt[64][72];         // V transposed: [d][n]
  __shared__ __align__(16) f16 Ps[4][16][72];      // per-wave P tile
  const int tid  = threadIdx.x;
  const int wave = tid >> 5, lane = tid & 31;
  const int hi = lane >> 4, lm = lane & 15;
  const int qb = blockIdx.x, h = blockIdx.y, b = blockIdx.z;
  const size_t rowbase = (size_t)b * SEQ + (size_t)qb * 64;
  const int qoff = h * DH, koff = DIM + h * DH, voff = 2 * DIM + h * DH;

#pragma unroll
  for (int t = 0; t < 4; ++t) {                    // load Q 64x64
    const int idx = tid + t * 128;
    const int r = idx >> 3, c8 = (idx & 7) * 8;
    *(v8h*)&Qs[r][c8] = *(const v8h*)&qkv[(rowbase + r) * (3 * DIM) + qoff + c8];
  }

  float mrow[8], lrow[8];
#pragma unroll
  for (int v = 0; v < 8; ++v) { mrow[v] = -1e30f; lrow[v] = 0.0f; }
  v8f ot[4] = {};
  const float scale = 0.125f;                      // 1/sqrt(64)

  for (int kb = 0; kb <= qb; ++kb) {
    const size_t kbase = (size_t)b * SEQ + (size_t)kb * 64;
    __syncthreads();
#pragma unroll
    for (int t = 0; t < 4; ++t) {                  // K block, row-major
      const int idx = tid + t * 128;
      const int r = idx >> 3, c8 = (idx & 7) * 8;
      *(v8h*)&Ks[r][c8] = *(const v8h*)&qkv[(kbase + r) * (3 * DIM) + koff + c8];
    }
    {                                              // V block, transposed into Vt
      const int r = tid >> 1, c0 = (tid & 1) * 32;
#pragma unroll
      for (int t = 0; t < 4; ++t) {
        v8h vv = *(const v8h*)&qkv[(kbase + r) * (3 * DIM) + voff + c0 + t * 8];
#pragma unroll
        for (int e = 0; e < 8; ++e) Vt[c0 + t * 8 + e][r] = vv[e];
      }
    }
    __syncthreads();

    v8f sacc[4] = {};                              // S = Q @ K^T (16x64 per wave)
#pragma unroll
    for (int ks = 0; ks < 2; ++ks) {
      const int ka = 8 * hi + 32 * ks;
      v16h a = mk16(*(const v8h*)&Qs[wave * 16 + lm][ka],
                    *(const v8h*)&Qs[wave * 16 + lm][ka + 16]);
      const int kk = 8 * hi + 32 * ks;
#pragma unroll
      for (int j = 0; j < 4; ++j) {
        v16h bf = mk16(*(const v8h*)&Ks[j * 16 + lm][kk],
                       *(const v8h*)&Ks[j * 16 + lm][kk + 16]);
        sacc[j] = wmma_f16(a, bf, sacc[j]);
      }
    }
    // online softmax: lane owns rows v + 8*hi, replicated over 16-lane groups
#pragma unroll
    for (int v = 0; v < 8; ++v) {
      float mx = -1e30f;
#pragma unroll
      for (int j = 0; j < 4; ++j) mx = fmaxf(mx, sacc[j][v] * scale);
#pragma unroll
      for (int off = 8; off > 0; off >>= 1) mx = fmaxf(mx, __shfl_xor(mx, off, 32));
      const float mnew = fmaxf(mrow[v], mx);
      const float corr = expf(mrow[v] - mnew);
      float rsum = 0.0f;
#pragma unroll
      for (int j = 0; j < 4; ++j) {
        const float p = expf(sacc[j][v] * scale - mnew);
        Ps[wave][v + 8 * hi][j * 16 + lm] = (f16)p;
        rsum += p;
      }
#pragma unroll
      for (int off = 8; off > 0; off >>= 1) rsum += __shfl_xor(rsum, off, 32);
      lrow[v] = lrow[v] * corr + rsum;
      mrow[v] = mnew;
#pragma unroll
      for (int j = 0; j < 4; ++j) ot[j][v] *= corr;
    }
    asm volatile("" ::: "memory");                 // keep LDS P store before reload
#pragma unroll
    for (int ks = 0; ks < 2; ++ks) {               // O += P @ V
      const int ka = 8 * hi + 32 * ks;
      v16h a = mk16(*(const v8h*)&Ps[wave][lm][ka],
                    *(const v8h*)&Ps[wave][lm][ka + 16]);
      const int kk = 8 * hi + 32 * ks;
#pragma unroll
      for (int j = 0; j < 4; ++j) {
        v16h bf = mk16(*(const v8h*)&Vt[j * 16 + lm][kk],
                       *(const v8h*)&Vt[j * 16 + lm][kk + 16]);
        ot[j] = wmma_f16(a, bf, ot[j]);
      }
    }
  }
#pragma unroll
  for (int j = 0; j < 4; ++j)
#pragma unroll
    for (int v = 0; v < 8; ++v) {
      const size_t row = rowbase + wave * 16 + v + 8 * hi;
      y[row * DIM + h * DH + j * 16 + lm] = (f16)(ot[j][v] / lrow[v]);
    }
}

// ---------------------------------------------------------------------------
extern "C" void kernel_launch(void* const* d_in, const int* in_sizes, int n_in,
                              void* d_out, int out_size, void* d_ws, size_t ws_size,
                              hipStream_t stream) {
  (void)in_sizes; (void)n_in; (void)out_size; (void)ws_size;
  const float* seq   = (const float*)d_in[0];
  const float* ln1w  = (const float*)d_in[1];
  const float* ln1b  = (const float*)d_in[2];
  const float* wqkv  = (const float*)d_in[3];
  const float* bqkv  = (const float*)d_in[4];
  const float* wo    = (const float*)d_in[5];
  const float* bo    = (const float*)d_in[6];
  const float* ln2w  = (const float*)d_in[7];
  const float* ln2b  = (const float*)d_in[8];
  const float* wfc   = (const float*)d_in[9];
  const float* bfc   = (const float*)d_in[10];
  const float* wproj = (const float*)d_in[11];
  const float* bproj = (const float*)d_in[12];
  const float* lnfw  = (const float*)d_in[13];
  const float* lnfb  = (const float*)d_in[14];

  char* ws = (char*)d_ws;
  size_t off = 0;
  auto alloc = [&](size_t bytes) -> char* {
    char* p = ws + off;
    off = (off + bytes + 255) & ~(size_t)255;
    return p;
  };
  f16* wqkv_t  = (f16*)alloc((size_t)LAYERS * 3 * DIM * DIM * sizeof(f16));
  f16* wo_t    = (f16*)alloc((size_t)LAYERS * DIM * DIM * sizeof(f16));
  f16* wfc_t   = (f16*)alloc((size_t)LAYERS * 4 * DIM * DIM * sizeof(f16));
  f16* wproj_t = (f16*)alloc((size_t)LAYERS * DIM * 4 * DIM * sizeof(f16));
  float* X     = (float*)alloc((size_t)ROWS * DIM * sizeof(float));
  f16* h16     = (f16*)alloc((size_t)ROWS * DIM * sizeof(f16));
  f16* qkv16   = (f16*)alloc((size_t)ROWS * 3 * DIM * sizeof(f16));
  f16* y16     = (f16*)alloc((size_t)ROWS * DIM * sizeof(f16));
  f16* u16     = (f16*)alloc((size_t)ROWS * 4 * DIM * sizeof(f16));

  dim3 tb(32, 8);
  wprep_kernel<<<dim3(3 * DIM / 32, DIM / 32, LAYERS), tb, 0, stream>>>(wqkv, wqkv_t, DIM, 3 * DIM);
  wprep_kernel<<<dim3(DIM / 32, DIM / 32, LAYERS),     tb, 0, stream>>>(wo, wo_t, DIM, DIM);
  wprep_kernel<<<dim3(4 * DIM / 32, DIM / 32, LAYERS), tb, 0, stream>>>(wfc, wfc_t, DIM, 4 * DIM);
  wprep_kernel<<<dim3(DIM / 32, 4 * DIM / 32, LAYERS), tb, 0, stream>>>(wproj, wproj_t, 4 * DIM, DIM);

  (void)hipMemcpyAsync(X, seq, (size_t)ROWS * DIM * sizeof(float),
                       hipMemcpyDeviceToDevice, stream);

  for (int l = 0; l < LAYERS; ++l) {
    ln_kernel<f16><<<ROWS, 256, 0, stream>>>(X, ln1w + l * DIM, ln1b + l * DIM, h16);
    gemm_kernel<EPI_BIAS, f16><<<dim3(ROWS / 128, 3 * DIM / 64), 256, 0, stream>>>(
        h16, wqkv_t + (size_t)l * 3 * DIM * DIM, bqkv + (size_t)l * 3 * DIM, nullptr,
        qkv16, ROWS, 3 * DIM, DIM);
    attn_kernel<<<dim3(SEQ / 64, HEADS, BATCH), 128, 0, stream>>>(qkv16, y16);
    gemm_kernel<EPI_RES, float><<<dim3(ROWS / 128, DIM / 64), 256, 0, stream>>>(
        y16, wo_t + (size_t)l * DIM * DIM, bo + (size_t)l * DIM, X, X, ROWS, DIM, DIM);
    ln_kernel<f16><<<ROWS, 256, 0, stream>>>(X, ln2w + l * DIM, ln2b + l * DIM, h16);
    gemm_kernel<EPI_GELU, f16><<<dim3(ROWS / 128, 4 * DIM / 64), 256, 0, stream>>>(
        h16, wfc_t + (size_t)l * 4 * DIM * DIM, bfc + (size_t)l * 4 * DIM, nullptr,
        u16, ROWS, 4 * DIM, DIM);
    gemm_kernel<EPI_RES, float><<<dim3(ROWS / 128, DIM / 64), 256, 0, stream>>>(
        u16, wproj_t + (size_t)l * DIM * 4 * DIM, bproj + (size_t)l * DIM, X, X,
        ROWS, DIM, 4 * DIM);
  }
  ln_kernel<float><<<ROWS, 256, 0, stream>>>(X, lnfw, lnfb, (float*)d_out);
}
